// GraphAttNet_87136296501510
// MI455X (gfx1250) — matrix operand order repaired and checked by reading, not compiled, not run on hardware
//
#include <hip/hip_runtime.h>

#define NN      50000
#define NE      800000
#define ETOT    (NE + NN)
#define LATENT  128
#define INFEAT  64
#define OUTG    32
#define NGRAPH  50
#define NPG     1000

typedef __attribute__((ext_vector_type(16))) __bf16 v16bf;
typedef __attribute__((ext_vector_type(8)))  float  v8f;

// Hardware f32 -> bf16 (RNE); backend selects v_cvt_(pk_)bf16_f32.
static __device__ __forceinline__ __bf16 f2bf(float f) { return (__bf16)f; }

// Pre-pack W[K x 128] (f32, row-major) into WMMA B-fragment lane order, bf16.
// One thread == one lane of one (nt, kt) fragment; writes 8 contiguous dwords.
// Layout: P[ (nt*ksteps + kt)*32*8 + lane*8 + v ] = pack(W[k,n], W[k+1,n])
__global__ __launch_bounds__(256)
void pack_b_kernel(const float* __restrict__ W, unsigned* __restrict__ P, int ksteps)
{
    int idx = blockIdx.x * blockDim.x + threadIdx.x;
    if (idx >= 8 * ksteps * 32) return;
    int lane = idx & 31;
    int frag = idx >> 5;               // nt*ksteps + kt
    int kt   = frag % ksteps;
    int nt   = frag / ksteps;
    int n    = nt * 16 + (lane & 15);
    int koff = (lane >> 4) * 16 + kt * 32;   // lanes 16-31 hold K half +16
    unsigned* dst = P + (size_t)frag * 32 * 8 + lane * 8;
#pragma unroll
    for (int v = 0; v < 8; ++v) {
        unsigned short lo = __builtin_bit_cast(unsigned short, f2bf(W[(size_t)(koff + 2 * v)     * 128 + n]));
        unsigned short hi = __builtin_bit_cast(unsigned short, f2bf(W[(size_t)(koff + 2 * v + 1) * 128 + n]));
        dst[v] = (unsigned)lo | ((unsigned)hi << 16);
    }
}

// C[M x 128] = act(A[M x (KSTEPS*32)] @ W + bias); W pre-packed bf16 B-fragments.
// One wave -> one 16-row M tile, all 8 N tiles. M multiple of 16.
// KSTEPS is compile-time so afr[] is statically indexed (no v_movrels).
template <int KSTEPS, int RELU>
__global__ __launch_bounds__(256)
void gemm_bf16_wmma(const float* __restrict__ A, const unsigned* __restrict__ PB,
                    const float* __restrict__ bias, float* __restrict__ C,
                    int M)
{
    constexpr int K = KSTEPS * 32;
    const int lane  = threadIdx.x & 31;
    const int wave  = threadIdx.x >> 5;
    const int mtile = blockIdx.x * 8 + wave;
    if (mtile * 16 >= M) return;                 // wave-uniform: EXEC stays all-1s
    const int mbase = mtile * 16;
    const int mrow  = mbase + (lane & 15);
    const int koffA = (lane >> 4) * 16;          // A: lanes 16-31 take K+16 window

    v16bf afr[KSTEPS];
#pragma unroll
    for (int kt = 0; kt < KSTEPS; ++kt) {
        const float* arow = A + (size_t)mrow * K + kt * 32 + koffA;
        v16bf a;
#pragma unroll
        for (int v = 0; v < 8; ++v) {
            int kb = 2 * v;
            a[2 * v]     = f2bf(arow[kb]);
            a[2 * v + 1] = f2bf(arow[kb + 1]);
        }
        afr[kt] = a;
    }

#pragma unroll
    for (int nt = 0; nt < 8; ++nt) {
        v8f acc = {};
#pragma unroll
        for (int kt = 0; kt < KSTEPS; ++kt) {
            const v16bf b = *(const v16bf*)(PB + ((size_t)(nt * KSTEPS + kt) * 32 + lane) * 8);
            acc = __builtin_amdgcn_wmma_f32_16x16x32_bf16(
                false, afr[kt], false, b, (short)0, acc, false, false);
        }
        const int mloc = (lane >> 4) << 3;       // lanes 16-31 hold M=v+8
        const int col  = nt * 16 + (lane & 15);
        const float bi = bias[col];
#pragma unroll
        for (int v = 0; v < 8; ++v) {
            float val = acc[v] + bi;
            if constexpr (RELU) val = fmaxf(val, 0.f);
            C[(size_t)(mbase + mloc + v) * 128 + col] = val;
        }
    }
}

// a_s[n] = h[n] . att_w[0:128], a_r[n] = h[n] . att_w[128:256]  (one wave/node)
__global__ __launch_bounds__(256)
void att_node_kernel(const float* __restrict__ h, const float* __restrict__ attw,
                     float* __restrict__ a_s, float* __restrict__ a_r)
{
    int node = blockIdx.x * 8 + (threadIdx.x >> 5);
    if (node >= NN) return;
    int lane = threadIdx.x & 31;
    const float* hp = h + (size_t)node * 128;
    float ss = 0.f, sr = 0.f;
#pragma unroll
    for (int j = 0; j < 4; ++j) {
        int i = lane + j * 32;
        float v = hp[i];
        ss += v * attw[i];
        sr += v * attw[128 + i];
    }
    for (int off = 16; off > 0; off >>= 1) {
        ss += __shfl_xor(ss, off, 32);
        sr += __shfl_xor(sr, off, 32);
    }
    if (lane == 0) { a_s[node] = ss; a_r[node] = sr; }
}

__global__ void init_seg(unsigned* __restrict__ segmax, float* __restrict__ segsum,
                         float* __restrict__ agg)
{
    size_t i = (size_t)blockIdx.x * blockDim.x + threadIdx.x;
    if (i < (size_t)NN * 128) agg[i] = 0.f;
    if (i < NN) { segmax[i] = 0u; segsum[i] = 0.f; }
}

static __device__ __forceinline__ unsigned f2ord(float f) {
    unsigned b = __builtin_bit_cast(unsigned, f);
    return (b & 0x80000000u) ? ~b : (b | 0x80000000u);
}
static __device__ __forceinline__ float ord2f(unsigned u) {
    unsigned b = (u & 0x80000000u) ? (u & 0x7FFFFFFFu) : ~u;
    return __builtin_bit_cast(float, b);
}

__global__ void edge_logit_max(const int* __restrict__ senders, const int* __restrict__ receivers,
                               const float* __restrict__ a_s, const float* __restrict__ a_r,
                               float* __restrict__ logits, unsigned* __restrict__ segmax)
{
    int e = blockIdx.x * blockDim.x + threadIdx.x;
    if (e >= ETOT) return;
    int s = (e < NE) ? senders[e]   : (e - NE);
    int r = (e < NE) ? receivers[e] : (e - NE);
    float lg = a_s[s] + a_r[r];        // att_b cancels in segment softmax
    logits[e] = lg;
    atomicMax(&segmax[r], f2ord(lg));
}

__global__ void edge_expsum(const int* __restrict__ receivers,
                            float* __restrict__ wbuf,                 // in: logits, out: exp
                            const unsigned* __restrict__ segmax,
                            float* __restrict__ segsum)
{
    int e = blockIdx.x * blockDim.x + threadIdx.x;
    if (e >= ETOT) return;
    int r = (e < NE) ? receivers[e] : (e - NE);
    float w = __expf(wbuf[e] - ord2f(segmax[r]));
    wbuf[e] = w;
    atomicAdd(&segsum[r], w);
}

// agg[recv] += (w/segsum[recv]) * h[send]   — one wave per edge, float4 per lane
__global__ __launch_bounds__(256)
void edge_aggregate(const int* __restrict__ senders, const int* __restrict__ receivers,
                    const float* __restrict__ h, const float* __restrict__ wbuf,
                    const float* __restrict__ segsum, float* __restrict__ agg)
{
    int e = blockIdx.x * 8 + (threadIdx.x >> 5);
    if (e >= ETOT) return;
    int lane = threadIdx.x & 31;
    int s = (e < NE) ? senders[e]   : (e - NE);
    int r = (e < NE) ? receivers[e] : (e - NE);
    float coef = wbuf[e] / segsum[r];
    float4 hv = ((const float4*)(h + (size_t)s * 128))[lane];
    float* ap = agg + (size_t)r * 128 + lane * 4;
    atomicAdd(ap + 0, coef * hv.x);
    atomicAdd(ap + 1, coef * hv.y);
    atomicAdd(ap + 2, coef * hv.z);
    atomicAdd(ap + 3, coef * hv.w);
}

// x = LayerNorm(leaky_relu(agg) + x) * scale + offset   — one wave per node
__global__ __launch_bounds__(256)
void node_update(const float* __restrict__ agg, float* __restrict__ x,
                 const float* __restrict__ lns, const float* __restrict__ lno)
{
    int node = blockIdx.x * 8 + (threadIdx.x >> 5);
    if (node >= NN) return;
    int lane = threadIdx.x & 31;
    float4 a  = ((const float4*)(agg + (size_t)node * 128))[lane];
    float4 xv = ((const float4*)(x   + (size_t)node * 128))[lane];
    float v[4];
    v[0] = (a.x > 0.f ? a.x : 0.01f * a.x) + xv.x;
    v[1] = (a.y > 0.f ? a.y : 0.01f * a.y) + xv.y;
    v[2] = (a.z > 0.f ? a.z : 0.01f * a.z) + xv.z;
    v[3] = (a.w > 0.f ? a.w : 0.01f * a.w) + xv.w;
    float sum = v[0] + v[1] + v[2] + v[3];
    float sq  = v[0]*v[0] + v[1]*v[1] + v[2]*v[2] + v[3]*v[3];
    for (int off = 16; off > 0; off >>= 1) {
        sum += __shfl_xor(sum, off, 32);
        sq  += __shfl_xor(sq,  off, 32);
    }
    float mu  = sum * (1.0f / 128.0f);
    float var = sq * (1.0f / 128.0f) - mu * mu;
    float inv = rsqrtf(var + 1e-5f);
    float4 o;
    int f = lane * 4;
    o.x = (v[0] - mu) * inv * lns[f + 0] + lno[f + 0];
    o.y = (v[1] - mu) * inv * lns[f + 1] + lno[f + 1];
    o.z = (v[2] - mu) * inv * lns[f + 2] + lno[f + 2];
    o.w = (v[3] - mu) * inv * lns[f + 3] + lno[f + 3];
    ((float4*)(x + (size_t)node * 128))[lane] = o;
}

__global__ __launch_bounds__(128)
void pool_kernel(const float* __restrict__ x, float* __restrict__ pooled)
{
    int g = blockIdx.x;
    int f = threadIdx.x;
    const float* xp = x + (size_t)g * NPG * 128 + f;
    float s = 0.f;
    for (int i = 0; i < NPG; ++i) s += xp[(size_t)i * 128];
    pooled[g * 128 + f] = s * (1.0f / NPG);
}

__global__ __launch_bounds__(32)
void decode_kernel(const float* __restrict__ pooled, const float* __restrict__ dw,
                   const float* __restrict__ db, float* __restrict__ out)
{
    int g = blockIdx.x;
    int o = threadIdx.x;
    const float* pp = pooled + g * 128;
    float s = db[o];
    for (int i = 0; i < 128; ++i) s += pp[i] * dw[i * 32 + o];
    out[g * 32 + o] = s;
}

extern "C" void kernel_launch(void* const* d_in, const int* in_sizes, int n_in,
                              void* d_out, int out_size, void* d_ws, size_t ws_size,
                              hipStream_t stream) {
    (void)in_sizes; (void)n_in; (void)out_size; (void)ws_size;
    const float* nodes     = (const float*)d_in[0];
    const int*   senders   = (const int*)d_in[3];
    const int*   receivers = (const int*)d_in[4];
    const float* emb_w     = (const float*)d_in[5];
    const float* emb_b     = (const float*)d_in[6];
    const float* mlp_w     = (const float*)d_in[7];
    const float* mlp_b     = (const float*)d_in[8];
    const float* att_w     = (const float*)d_in[9];
    const float* ln_scale  = (const float*)d_in[11];
    const float* ln_offset = (const float*)d_in[12];
    const float* dec_w     = (const float*)d_in[13];
    const float* dec_b     = (const float*)d_in[14];
    float* out = (float*)d_out;

    char* ws = (char*)d_ws;
    size_t off = 0;
    auto alloc = [&](size_t bytes) -> void* {
        void* p = ws + off;
        off = (off + bytes + 255) & ~(size_t)255;
        return p;
    };
    float*    x      = (float*)alloc((size_t)NN * 128 * 4);
    float*    h      = (float*)alloc((size_t)NN * 128 * 4);
    float*    t      = (float*)alloc((size_t)NN * 128 * 4);   // MLP hidden, reused as agg
    float*    a_s    = (float*)alloc((size_t)NN * 4);
    float*    a_r    = (float*)alloc((size_t)NN * 4);
    unsigned* segmax = (unsigned*)alloc((size_t)NN * 4);
    float*    segsum = (float*)alloc((size_t)NN * 4);
    float*    wbuf   = (float*)alloc((size_t)ETOT * 4);
    float*    pooled = (float*)alloc((size_t)NGRAPH * 128 * 4);
    unsigned* pemb   = (unsigned*)alloc((size_t)8 * 2 * 32 * 8 * 4);   // 16 KB
    unsigned* pmlp[4];
    for (int i = 0; i < 4; ++i)
        pmlp[i] = (unsigned*)alloc((size_t)8 * 4 * 32 * 8 * 4);        // 32 KB each

    const int gemmBlocks = (NN / 16 + 7) / 8;          // 3125 M-tiles, 8 waves/block
    const int edgeBlocks = (ETOT + 255) / 256;
    const int edgeWaveBlocks = (ETOT + 7) / 8;
    const int nodeWaveBlocks = (NN + 7) / 8;
    const int initBlocks = ((NN * 128) + 255) / 256;

    // Pre-pack all weight matrices into bf16 WMMA B-fragment layout (once).
    pack_b_kernel<<<2, 256, 0, stream>>>(emb_w, pemb, 2);
    for (int i = 0; i < 4; ++i)
        pack_b_kernel<<<4, 256, 0, stream>>>(mlp_w + (size_t)i * 128 * 128, pmlp[i], 4);

    // embed: x = nodes @ emb_w + emb_b
    gemm_bf16_wmma<2, 0><<<gemmBlocks, 256, 0, stream>>>(nodes, pemb, emb_b, x, NN);

    for (int s = 0; s < 2; ++s) {
        const float* b0 = mlp_b + (size_t)(s * 2 + 0) * 128;
        const float* b1 = mlp_b + (size_t)(s * 2 + 1) * 128;
        // h = relu(relu(x @ w0 + b0) @ w1 + b1)
        gemm_bf16_wmma<4, 1><<<gemmBlocks, 256, 0, stream>>>(x, pmlp[s * 2 + 0], b0, t, NN);
        gemm_bf16_wmma<4, 1><<<gemmBlocks, 256, 0, stream>>>(t, pmlp[s * 2 + 1], b1, h, NN);
        // per-node attention dots (factored logit GEMV)
        att_node_kernel<<<nodeWaveBlocks, 256, 0, stream>>>(h, att_w + (size_t)s * 256, a_s, a_r);
        // segment softmax + weighted scatter (agg reuses t)
        init_seg<<<initBlocks, 256, 0, stream>>>(segmax, segsum, t);
        edge_logit_max<<<edgeBlocks, 256, 0, stream>>>(senders, receivers, a_s, a_r, wbuf, segmax);
        edge_expsum<<<edgeBlocks, 256, 0, stream>>>(receivers, wbuf, segmax, segsum);
        edge_aggregate<<<edgeWaveBlocks, 256, 0, stream>>>(senders, receivers, h, wbuf, segsum, t);
        // skip + leaky_relu + layernorm, in-place on x
        node_update<<<nodeWaveBlocks, 256, 0, stream>>>(t, x,
            ln_scale + (size_t)s * 128, ln_offset + (size_t)s * 128);
    }

    pool_kernel<<<NGRAPH, 128, 0, stream>>>(x, pooled);
    decode_kernel<<<NGRAPH, 32, 0, stream>>>(pooled, dec_w, dec_b, out);
}
